// Four_Headed_Attention_64287070486549
// MI455X (gfx1250) — matrix operand order repaired and checked by reading, not compiled, not run on hardware
//
#include <hip/hip_runtime.h>

// Problem constants (match reference)
#define BB 4
#define SS 2048
#define EE 1024
#define HH 4
#define DD 256   // head dim
#define NEG_BIG (-1e30f)

typedef __attribute__((ext_vector_type(16))) __bf16        v16bf;
typedef __attribute__((ext_vector_type(8)))  float         v8f;
typedef __attribute__((ext_vector_type(4)))  unsigned int  v4u;
typedef __attribute__((ext_vector_type(4)))  float         v4f;

union Frag {
    v16bf          v;      // WMMA operand (16 bf16 per lane = 8 VGPRs)
    v4u            q[2];   // two 16-byte chunks
    __bf16         b[16];
    unsigned short u[16];
};

// Hardware bf16 convert (RTNE) -> v_cvt_pk_bf16_f32
__device__ __forceinline__ unsigned short bf16_bits(float f) {
    union { __bf16 h; unsigned short u; } x;
    x.h = (__bf16)f;
    return x.u;
}

__device__ __forceinline__ v8f wmma_bf16(v16bf a, v16bf b, v8f c) {
    // v_wmma_f32_16x16x32_bf16, f32 accumulate
    return __builtin_amdgcn_wmma_f32_16x16x32_bf16(false, a, false, b, (short)0, c, false, false);
}

// reduce across the 16-lane group (matches C-matrix column striping)
__device__ __forceinline__ float rowmax16(float v) {
    v = fmaxf(v, __shfl_xor(v, 1, 32));
    v = fmaxf(v, __shfl_xor(v, 2, 32));
    v = fmaxf(v, __shfl_xor(v, 4, 32));
    v = fmaxf(v, __shfl_xor(v, 8, 32));
    return v;
}
__device__ __forceinline__ float rowsum16(float v) {
    v += __shfl_xor(v, 1, 32);
    v += __shfl_xor(v, 2, 32);
    v += __shfl_xor(v, 4, 32);
    v += __shfl_xor(v, 8, 32);
    return v;
}

// ---------------------------------------------------------------------------
// Weight prep: W[g][e][d] (f32) -> Wt[g][d][e] (bf16)
// ---------------------------------------------------------------------------
__global__ void prepw_kernel(const float* __restrict__ W,
                             unsigned short* __restrict__ Wt,
                             int Edim, int Ddim) {
    int g = blockIdx.y;
    long long i = (long long)blockIdx.x * blockDim.x + threadIdx.x;
    long long n = (long long)Edim * Ddim;
    if (i >= n) return;
    int e = (int)(i / Ddim);
    int d = (int)(i % Ddim);
    Wt[((size_t)g * Ddim + d) * Edim + e] = bf16_bits(W[(size_t)g * n + i]);
}

// ---------------------------------------------------------------------------
// Projection: out[b,h,s,d] = sum_e X[b,s,e] * W[h,e,d] + bias[h,d]
// 16(s) x 64(d) tile per wave; A built once per k-step, 4 B-frags loaded as
// one clause, then 4 WMMAs (ILP: loads cover WMMA latency via partial waits).
// ---------------------------------------------------------------------------
__global__ __launch_bounds__(32)
void proj_kernel(const float* __restrict__ X,            // [B,S,E] f32
                 const unsigned short* __restrict__ Wt,  // [H,D,E] bf16
                 const float* __restrict__ bias,         // [H,D]
                 unsigned short* __restrict__ Out,       // bf16
                 int transpose_out) {
    const int lane = threadIdx.x;
    const int n      = lane & 15;
    const int hi     = lane >> 4;
    const int h8     = hi * 8;
    const int half16 = hi * 16;

    const int stile = blockIdx.x;        // S/16
    const int dt0   = blockIdx.y * 4;    // first of 4 d-tiles (D/64 groups)
    const int bh    = blockIdx.z;        // B*H
    const int b = bh >> 2, h = bh & 3;

    const float* Xrow =
        X + ((size_t)b * SS + stile * 16 + n) * EE;               // A row m=n
    const unsigned short* Wrow =
        Wt + ((size_t)h * DD + dt0 * 16 + n) * EE;                // B col = n

    v8f acc[4];
#pragma unroll
    for (int j = 0; j < 4; ++j) acc[j] = (v8f){};

#pragma unroll 2
    for (int c = 0; c < EE / 32; ++c) {
        // A-fragment: f32 -> bf16 via HW convert (two 32B chunks, ISA layout)
        Frag a;
        const v4f* x0 = (const v4f*)(Xrow + c * 32 + h8);
        const v4f* x1 = (const v4f*)(Xrow + c * 32 + 16 + h8);
        v4f f0 = x0[0], f1 = x0[1], f2 = x1[0], f3 = x1[1];
        // 4 B-fragments: issue all 8 b128 loads up front (one clause)
        Frag bfr[4];
#pragma unroll
        for (int j = 0; j < 4; ++j) {
            const unsigned short* wp =
                Wrow + (size_t)j * 16 * EE + c * 32 + h8;
            bfr[j].q[0] = *(const v4u*)(wp);
            bfr[j].q[1] = *(const v4u*)(wp + 16);
        }
#pragma unroll
        for (int i = 0; i < 4; ++i) {
            a.b[i]      = (__bf16)f0[i];
            a.b[4 + i]  = (__bf16)f1[i];
            a.b[8 + i]  = (__bf16)f2[i];
            a.b[12 + i] = (__bf16)f3[i];
        }
#pragma unroll
        for (int j = 0; j < 4; ++j) acc[j] = wmma_bf16(a.v, bfr[j].v, acc[j]);
    }

#pragma unroll
    for (int j = 0; j < 4; ++j) {
        const int dcol = (dt0 + j) * 16 + n;
        const float bv = bias[h * DD + dcol];
        if (!transpose_out) {
            // row-major [b,h,s,d]
            unsigned short* op =
                Out + (((size_t)bh * SS) + stile * 16) * DD + dcol;
#pragma unroll
            for (int r = 0; r < 8; ++r)
                op[(size_t)(r + h8) * DD] = bf16_bits(acc[j][r] + bv);
        } else {
            // Vt[b,h,d,s]: 8 consecutive s per lane -> one 16B store
            union { v4u q; __bf16 v[8]; } pk;
#pragma unroll
            for (int r = 0; r < 8; ++r) pk.v[r] = (__bf16)(acc[j][r] + bv);
            unsigned short* op =
                Out + (((size_t)bh * DD) + dcol) * SS + stile * 16 + h8;
            *(v4u*)op = pk.q;
        }
    }
}

// ---------------------------------------------------------------------------
// Fused causal attention, flash-style. One wave per (b,h, 16-row q tile).
// ---------------------------------------------------------------------------
__global__ __launch_bounds__(32)
void attn_kernel(const unsigned short* __restrict__ Qh,  // [B,H,S,D] bf16
                 const unsigned short* __restrict__ Kh,  // [B,H,S,D] bf16
                 const unsigned short* __restrict__ Vt,  // [B,H,D,S] bf16
                 unsigned short* __restrict__ O) {       // [B,S,E]  bf16
    __shared__ __align__(16) unsigned short p_lds[16 * 32];

    const int lane = threadIdx.x;
    const int n      = lane & 15;
    const int hi     = lane >> 4;
    const int h8     = hi * 8;
    const int half16 = hi * 16;

    const int qt = blockIdx.x;           // S/16
    const int bh = blockIdx.y;           // B*H
    const int b = bh >> 2, h = bh & 3;
    const int qbase = qt * 16;

    // Preload Q A-fragments for all 8 K-chunks of Dh=256
    Frag qf[8];
    const unsigned short* Qrow = Qh + ((size_t)bh * SS + qbase + n) * DD;
#pragma unroll
    for (int c = 0; c < 8; ++c) {
        qf[c].q[0] = *(const v4u*)(Qrow + c * 32 + h8);
        qf[c].q[1] = *(const v4u*)(Qrow + c * 32 + 16 + h8);
    }

    v8f acc[16];
#pragma unroll
    for (int t = 0; t < 16; ++t) acc[t] = (v8f){};
    float M[8], L[8];
#pragma unroll
    for (int r = 0; r < 8; ++r) { M[r] = NEG_BIG; L[r] = 0.0f; }

    const float scale = 1.0f / (float)DD;  // reference divides by head_dim
    const int nkb = (qbase + 47) / 32;     // causal: cover k <= qbase+15

    for (int kb = 0; kb < nkb; ++kb) {
        const int k0 = kb * 32;
        const unsigned short* Krow0 = Kh + ((size_t)bh * SS + k0 + n) * DD;
        const unsigned short* Krow1 = Krow0 + (size_t)16 * DD;
        const unsigned short* Vbase =
            Vt + ((size_t)bh * DD + n) * SS + k0 + h8;
        if (kb + 1 < nkb)
            __builtin_prefetch(Krow0 + (size_t)32 * DD, 0, 0);  // next K block

        // ---- scores: two 16x16 sub-tiles over 32 k columns ----
        v8f s0 = {}, s1 = {};
#pragma unroll
        for (int c = 0; c < 8; ++c) {
            Frag kf0, kf1;   // both loads issued before the WMMAs
            kf0.q[0] = *(const v4u*)(Krow0 + c * 32 + h8);
            kf0.q[1] = *(const v4u*)(Krow0 + c * 32 + 16 + h8);
            kf1.q[0] = *(const v4u*)(Krow1 + c * 32 + h8);
            kf1.q[1] = *(const v4u*)(Krow1 + c * 32 + 16 + h8);
            s0 = wmma_bf16(qf[c].v, kf0.v, s0);
            s1 = wmma_bf16(qf[c].v, kf1.v, s1);
        }

        // ---- online softmax update ----
        float alpha[8], p0[8], p1[8];
#pragma unroll
        for (int r = 0; r < 8; ++r) {
            const int qrow = qbase + h8 + r;
            float a0 = s0[r] * scale;
            float a1 = s1[r] * scale;
            if (k0 + n > qrow)      a0 = NEG_BIG;   // strict upper-tri mask
            if (k0 + 16 + n > qrow) a1 = NEG_BIG;
            float mx = rowmax16(fmaxf(a0, a1));
            float Mn = fmaxf(M[r], mx);
            alpha[r] = __expf(M[r] - Mn);
            float e0 = (a0 <= -1e29f) ? 0.0f : __expf(a0 - Mn);
            float e1 = (a1 <= -1e29f) ? 0.0f : __expf(a1 - Mn);
            L[r] = L[r] * alpha[r] + rowsum16(e0 + e1);
            M[r] = Mn;
            p0[r] = e0;
            p1[r] = e1;
        }
#pragma unroll
        for (int t = 0; t < 16; ++t)
#pragma unroll
            for (int r = 0; r < 8; ++r) acc[t][r] *= alpha[r];

        // Prefetch V group 0 now: independent of the P LDS round-trip, so
        // these loads overlap the softmax/LDS serial chain.
        Frag vcur[4], vnext[4];
#pragma unroll
        for (int j = 0; j < 4; ++j) {
            const unsigned short* vp = Vbase + (size_t)j * 16 * SS;
            vcur[j].q[0] = *(const v4u*)(vp);
            vcur[j].q[1] = *(const v4u*)(vp + 16);
        }

        // ---- C-layout P -> A-layout via LDS (16x32 bf16 tile, 1 KB) ----
#pragma unroll
        for (int r = 0; r < 8; ++r) {
            p_lds[(r + h8) * 32 + n]      = bf16_bits(p0[r]);
            p_lds[(r + h8) * 32 + 16 + n] = bf16_bits(p1[r]);
        }
        __syncthreads();
        Frag pf;
        const unsigned short* pl = p_lds + n * 32;
        pf.q[0] = *(const v4u*)(pl + h8);
        pf.q[1] = *(const v4u*)(pl + 16 + h8);
        __syncthreads();  // WAR guard before next iteration's stores

        // ---- P (16x32) x V (32x16): 16 d-tiles, double-buffered groups of 4
#pragma unroll
        for (int tg = 0; tg < 4; ++tg) {
            if (tg < 3) {
#pragma unroll
                for (int j = 0; j < 4; ++j) {
                    const unsigned short* vp =
                        Vbase + (size_t)((tg + 1) * 4 + j) * 16 * SS;
                    vnext[j].q[0] = *(const v4u*)(vp);
                    vnext[j].q[1] = *(const v4u*)(vp + 16);
                }
            }
#pragma unroll
            for (int j = 0; j < 4; ++j)
                acc[tg * 4 + j] = wmma_bf16(pf.v, vcur[j].v, acc[tg * 4 + j]);
#pragma unroll
            for (int j = 0; j < 4; ++j) vcur[j] = vnext[j];  // renamed away
        }
    }

    // ---- normalize and store O[b, s, h*D + d] (heads concatenated) ----
#pragma unroll
    for (int r = 0; r < 8; ++r) {
        const float inv = 1.0f / L[r];
        unsigned short* orow =
            O + ((size_t)b * SS + qbase + h8 + r) * EE + h * DD + n;
#pragma unroll
        for (int t = 0; t < 16; ++t)
            orow[t * 16] = bf16_bits(acc[t][r] * inv);
    }
}

// ---------------------------------------------------------------------------
// Output projection: Y[r,f] = sum_e O[r,e] * Wo[e,f] + bo[f]   (f32 out)
// 16x64 tile per wave; B-frags loaded as one clause, then 4 WMMAs.
// ---------------------------------------------------------------------------
__global__ __launch_bounds__(32)
void outproj_kernel(const unsigned short* __restrict__ O,    // [B*S, E] bf16
                    const unsigned short* __restrict__ Wot,  // [E(f), E(e)] bf16
                    const float* __restrict__ bo,            // [E]
                    float* __restrict__ Y) {                 // [B*S, E] f32
    const int lane = threadIdx.x;
    const int n      = lane & 15;
    const int hi     = lane >> 4;
    const int h8     = hi * 8;
    const int half16 = hi * 16;

    const int rt  = blockIdx.x;        // (B*S)/16
    const int ft0 = blockIdx.y * 4;    // first of 4 f-tiles (E/64 groups)

    const unsigned short* Orow = O + ((size_t)rt * 16 + n) * EE;
    const unsigned short* Wrow = Wot + ((size_t)(ft0 * 16 + n)) * EE;

    v8f acc[4];
#pragma unroll
    for (int j = 0; j < 4; ++j) acc[j] = (v8f){};

#pragma unroll 2
    for (int c = 0; c < EE / 32; ++c) {
        Frag a;
        a.q[0] = *(const v4u*)(Orow + c * 32 + h8);
        a.q[1] = *(const v4u*)(Orow + c * 32 + 16 + h8);
        Frag bfr[4];
#pragma unroll
        for (int j = 0; j < 4; ++j) {
            const unsigned short* wp =
                Wrow + (size_t)j * 16 * EE + c * 32 + h8;
            bfr[j].q[0] = *(const v4u*)(wp);
            bfr[j].q[1] = *(const v4u*)(wp + 16);
        }
#pragma unroll
        for (int j = 0; j < 4; ++j) acc[j] = wmma_bf16(a.v, bfr[j].v, acc[j]);
    }
#pragma unroll
    for (int j = 0; j < 4; ++j) {
        const int fcol = (ft0 + j) * 16 + n;
        const float bv = bo[fcol];
        float* yp = Y + ((size_t)rt * 16 + h8) * EE + fcol;
#pragma unroll
        for (int r = 0; r < 8; ++r) yp[(size_t)r * EE] = acc[j][r] + bv;
    }
}

// ---------------------------------------------------------------------------
extern "C" void kernel_launch(void* const* d_in, const int* in_sizes, int n_in,
                              void* d_out, int out_size, void* d_ws, size_t ws_size,
                              hipStream_t stream) {
    const float* Xk = (const float*)d_in[0];
    const float* Xv = (const float*)d_in[1];
    const float* Xq = (const float*)d_in[2];
    const float* Wk = (const float*)d_in[3];
    const float* bk = (const float*)d_in[4];
    const float* Wq = (const float*)d_in[5];
    const float* bq = (const float*)d_in[6];
    const float* Wv = (const float*)d_in[7];
    const float* bv = (const float*)d_in[8];
    const float* Wo = (const float*)d_in[9];
    const float* bo = (const float*)d_in[10];
    float* Y = (float*)d_out;

    // Workspace carve-up (bf16 everywhere): ~72 MB total
    char* ws = (char*)d_ws;
    size_t off = 0;
    auto carve = [&](size_t bytes) {
        void* p = ws + off;
        off = (off + bytes + 255) & ~(size_t)255;
        return p;
    };
    const size_t NQKV = (size_t)BB * HH * SS * DD;  // 8M elements
    const size_t NO   = (size_t)BB * SS * EE;       // 8M elements
    const size_t NW   = (size_t)HH * EE * DD;       // 1M elements
    unsigned short* Qh  = (unsigned short*)carve(NQKV * 2);
    unsigned short* Kh  = (unsigned short*)carve(NQKV * 2);
    unsigned short* Vth = (unsigned short*)carve(NQKV * 2);
    unsigned short* Oc  = (unsigned short*)carve(NO * 2);
    unsigned short* Wqt = (unsigned short*)carve(NW * 2);
    unsigned short* Wkt = (unsigned short*)carve(NW * 2);
    unsigned short* Wvt = (unsigned short*)carve(NW * 2);
    unsigned short* Wot = (unsigned short*)carve((size_t)EE * EE * 2);

    // 1) weight convert + transpose to bf16
    {
        dim3 g((EE * DD + 255) / 256, HH);
        prepw_kernel<<<g, 256, 0, stream>>>(Wq, Wqt, EE, DD);
        prepw_kernel<<<g, 256, 0, stream>>>(Wk, Wkt, EE, DD);
        prepw_kernel<<<g, 256, 0, stream>>>(Wv, Wvt, EE, DD);
        dim3 go((EE * EE + 255) / 256, 1);
        prepw_kernel<<<go, 256, 0, stream>>>(Wo, Wot, EE, EE);
    }
    // 2) Q/K/V projections (V stored transposed); 16x64 tile per wave
    {
        dim3 g(SS / 16, DD / 64, BB * HH);
        proj_kernel<<<g, 32, 0, stream>>>(Xq, Wqt, bq, Qh, 0);
        proj_kernel<<<g, 32, 0, stream>>>(Xk, Wkt, bk, Kh, 0);
        proj_kernel<<<g, 32, 0, stream>>>(Xv, Wvt, bv, Vth, 1);
    }
    // 3) fused causal attention
    {
        dim3 g(SS / 16, BB * HH);
        attn_kernel<<<g, 32, 0, stream>>>(Qh, Kh, Vth, Oc);
    }
    // 4) output projection -> f32; 16x64 tile per wave
    {
        dim3 g((BB * SS) / 16, EE / 64);
        outproj_kernel<<<g, 32, 0, stream>>>(Oc, Wot, bo, Y);
    }
}
